// InteractionBlock_50843822850739
// MI455X (gfx1250) — matrix operand order repaired and checked by reading, not compiled, not run on hardware
//
#include <hip/hip_runtime.h>
#include <math.h>

#define NN  10000
#define EE  160000
#define CC  64
#define RBB 8
#define HH  64

typedef __attribute__((ext_vector_type(16))) _Float16 v16h;
typedef __attribute__((ext_vector_type(8)))  _Float16 v8h;
typedef __attribute__((ext_vector_type(8)))  float    v8f;

// D = A(16x32 f16) * B(32x16 f16) + C(16x16 f32), wave32 WMMA
__device__ __forceinline__ v8f wmma_f16(v16h a, v16h b, v8f c) {
  return __builtin_amdgcn_wmma_f32_16x16x32_f16(false, a, false, b, (short)0, c, false, false);
}

// A fragment: row-major [rows][lda] f16. ISA layout: lane L<16 -> row M=L, K=k0+0..7 (elts 0..7),
// K=k0+16..23 (elts 8..15); lanes 16..31 same rows with K base +8.
__device__ __forceinline__ v16h frag_a(const _Float16* A, int lda, int row0, int k0) {
  int lane = threadIdx.x & 31;
  const _Float16* p = A + (size_t)(row0 + (lane & 15)) * lda + k0 + ((lane >> 4) << 3);
  v8h lo = *(const v8h*)p;
  v8h hi = *(const v8h*)(p + 16);
  return __builtin_shufflevector(lo, hi, 0,1,2,3,4,5,6,7,8,9,10,11,12,13,14,15);
}

// B fragment from TRANSPOSED weights Bt[N][ldb=K] f16. Layout matches frag_a: lane L<16 -> K=k0+0..7 (elts 0..7),
// K=k0+16..23 (elts 8..15); lanes 16..31 K base +8, N = lane&15.
__device__ __forceinline__ v16h frag_b(const _Float16* Bt, int ldb, int col0, int k0) {
  int lane = threadIdx.x & 31;
  const _Float16* p = Bt + (size_t)(col0 + (lane & 15)) * ldb + k0 + ((lane >> 4) << 3);
  v8h lo = *(const v8h*)p;
  v8h hi = *(const v8h*)(p + 16);
  return __builtin_shufflevector(lo, hi, 0,1,2,3,4,5,6,7,8,9,10,11,12,13,14,15);
}

__global__ __launch_bounds__(256) void k_zero(float* __restrict__ p, long long n) {
  long long i = (long long)blockIdx.x * blockDim.x + threadIdx.x;
  long long s = (long long)gridDim.x * blockDim.x;
  for (; i < n; i += s) p[i] = 0.0f;
}

// Convert + transpose all weights to f16 (Bt[n][k] = W[k][n]) in global scratch.
__global__ __launch_bounds__(256) void k_prep(
    const float* __restrict__ Wup, const float* __restrict__ w1, const float* __restrict__ w2,
    const float* __restrict__ w3,  const float* __restrict__ w4, const float* __restrict__ Wdn,
    _Float16* __restrict__ wupt, _Float16* __restrict__ w1t, _Float16* __restrict__ w2t,
    _Float16* __restrict__ w3t,  _Float16* __restrict__ w4t, _Float16* __restrict__ wdnt) {
  int i = blockIdx.x * blockDim.x + threadIdx.x;
  int stride = gridDim.x * blockDim.x;
  for (int j = i; j < 4096; j += stride) wupt[j] = (_Float16)Wup[(j & 63) * 64 + (j >> 6)];
  for (int j = i; j < 2048; j += stride) {            // [64][32], K padded 8->32 with zeros
    int nn = j >> 5, k = j & 31;
    w1t[j] = (k < 8) ? (_Float16)w1[k * 64 + nn] : (_Float16)0.0f;
  }
  for (int j = i; j < 4096; j += stride) w2t[j] = (_Float16)w2[(j & 63) * 64 + (j >> 6)];
  for (int j = i; j < 4096; j += stride) w3t[j] = (_Float16)w3[(j & 63) * 64 + (j >> 6)];
  for (int j = i; j < 16384; j += stride) {           // [256][64]
    int nn = j >> 6, k = j & 63;
    w4t[j] = (_Float16)w4[k * 256 + nn];
  }
  for (int j = i; j < 16384; j += stride) {           // 4 x [64][64]
    int l = j >> 12, r = j & 4095;
    int dch = r >> 6, c = r & 63;
    wdnt[j] = (_Float16)Wdn[l * 4096 + c * 64 + dch];
  }
}

// xup = node_feats @ W_up, f16 WMMA, f32 out. 32 rows/block, 8 waves = 2x4 tiles.
__global__ __launch_bounds__(256) void k_linear_up(
    const float* __restrict__ nf, const _Float16* __restrict__ wupt, float* __restrict__ xup) {
  __shared__ _Float16 a_lds[32 * 64];
  int tid = threadIdx.x;
  int row0g = blockIdx.x * 32;
  for (int i = tid; i < 2048; i += 256) {
    int rr = i >> 6, cc = i & 63;
    int nr = row0g + rr;
    a_lds[i] = (_Float16)((nr < NN) ? nf[(size_t)nr * 64 + cc] : 0.0f);
  }
  __syncthreads();
  int wid = tid >> 5, rt = wid >> 2, ct = wid & 3;
  v8f acc = {};
  acc = wmma_f16(frag_a(a_lds, 64, rt * 16, 0),  frag_b(wupt, 64, ct * 16, 0),  acc);
  acc = wmma_f16(frag_a(a_lds, 64, rt * 16, 32), frag_b(wupt, 64, ct * 16, 32), acc);
  int lane = tid & 31, nc = lane & 15, rb = (lane >> 4) << 3;
  for (int r = 0; r < 8; ++r) {
    int nr = row0g + rt * 16 + rb + r;
    if (nr < NN) xup[(size_t)nr * 64 + ct * 16 + nc] = acc[r];
  }
}

// Fused: radial MLP (WMMA) -> mix, sph harmonics, gather sf, scatter-add messages.
// 128 edges/block, 8 waves x 16 edges each.
__global__ __launch_bounds__(256) void k_edges(
    const float* __restrict__ vectors, const float* __restrict__ radial,
    const int* __restrict__ senders, const int* __restrict__ receivers,
    const float* __restrict__ xup,
    const _Float16* __restrict__ w1t, const _Float16* __restrict__ w2t,
    const _Float16* __restrict__ w3t, const _Float16* __restrict__ w4t,
    float* __restrict__ accum) {
  __shared__ _Float16 re_h[128 * 32];   // radial embedding, K padded to 32
  __shared__ _Float16 h_a[128 * 64];    // hidden activations (reused h1->h2->h3)
  __shared__ float    y_lds[128 * 16];  // Y1(3) Y2(5) Y3(7) per edge
  __shared__ float    sf_lds[128 * 64]; // gathered sender features (reused 4x per irrep)
  __shared__ int      snd[128];
  __shared__ int      rcv[128];
  int tid = threadIdx.x;
  int e0 = blockIdx.x * 128;
  if (tid < 128) {
    int e = e0 + tid;
    float vx = vectors[3 * (size_t)e], vy = vectors[3 * (size_t)e + 1], vz = vectors[3 * (size_t)e + 2];
    float nrm = sqrtf(vx * vx + vy * vy + vz * vz) + 1e-12f;
    float inv = 1.0f / nrm;
    float x = vx * inv, y = vy * inv, z = vz * inv;
    const float s3 = 1.7320508075688772f, s5 = 2.23606797749979f, s15 = 3.872983346207417f;
    const float c1 = 2.0916500663351889f, c2 = 10.246950765959598f;
    const float c3 = 1.6201851746019651f, c4 = 1.3228756555322954f;
    float* yl = &y_lds[tid * 16];
    yl[0] = s3 * x; yl[1] = s3 * y; yl[2] = s3 * z;
    yl[3] = s15 * x * y; yl[4] = s15 * y * z; yl[5] = 0.5f * s5 * (3.0f * z * z - 1.0f);
    yl[6] = s15 * x * z; yl[7] = 0.5f * s15 * (x * x - y * y);
    yl[8]  = c1 * y * (3.0f * x * x - y * y);
    yl[9]  = c2 * x * y * z;
    yl[10] = c3 * y * (5.0f * z * z - 1.0f);
    yl[11] = c4 * z * (5.0f * z * z - 3.0f);
    yl[12] = c3 * x * (5.0f * z * z - 1.0f);
    yl[13] = 0.5f * c2 * z * (x * x - y * y);
    yl[14] = c1 * x * (x * x - y * y);
    yl[15] = 0.0f;
    int s = senders[e];
    snd[tid] = s;
    rcv[tid] = receivers[e];
    __builtin_prefetch(xup + (size_t)s * 64, 0, 1);   // warm sender row (global_prefetch)
    #pragma unroll
    for (int k = 0; k < 8; ++k)  re_h[tid * 32 + k] = (_Float16)radial[(size_t)e * 8 + k];
    #pragma unroll
    for (int k = 8; k < 32; ++k) re_h[tid * 32 + k] = (_Float16)0.0f;
  }
  __syncthreads();

  // stage sf = xup[senders] once per block (float4 -> global_load_b128), reused by all 4 irreps
  for (int i = tid; i < 128 * 16; i += 256) {
    int le = i >> 4, c4 = i & 15;
    const float4* src = (const float4*)(xup + (size_t)snd[le] * 64) + c4;
    ((float4*)sf_lds)[i] = *src;
  }
  __syncthreads();

  int wid = tid >> 5;
  int row0 = wid * 16;          // this wave's 16 edges within the block
  int lane = tid & 31;
  int n = lane & 15;
  int rb = (lane >> 4) << 3;
  v8f d[4];

  // layer 1: [16x32]@[32x64] (K padded), silu
  {
    v16h a = frag_a(re_h, 32, row0, 0);
    #pragma unroll
    for (int t = 0; t < 4; ++t) {
      v8f acc = {};
      d[t] = wmma_f16(a, frag_b(w1t, 32, t * 16, 0), acc);
    }
    #pragma unroll
    for (int t = 0; t < 4; ++t)
      for (int r = 0; r < 8; ++r) {
        float v = d[t][r];
        v = v / (1.0f + __expf(-v));
        h_a[(row0 + rb + r) * 64 + t * 16 + n] = (_Float16)v;
      }
  }
  // layers 2,3: [16x64]@[64x64], silu (in-order LDS within wave makes in-place safe)
  const _Float16* ws23[2] = {w2t, w3t};
  for (int L = 0; L < 2; ++L) {
    v16h a0 = frag_a(h_a, 64, row0, 0);
    v16h a1 = frag_a(h_a, 64, row0, 32);
    #pragma unroll
    for (int t = 0; t < 4; ++t) {
      v8f acc = {};
      acc = wmma_f16(a0, frag_b(ws23[L], 64, t * 16, 0),  acc);
      acc = wmma_f16(a1, frag_b(ws23[L], 64, t * 16, 32), acc);
      d[t] = acc;
    }
    #pragma unroll
    for (int t = 0; t < 4; ++t)
      for (int r = 0; r < 8; ++r) {
        float v = d[t][r];
        v = v / (1.0f + __expf(-v));
        h_a[(row0 + rb + r) * 64 + t * 16 + n] = (_Float16)v;
      }
  }
  // layer 4: [16x64]@[64x256] -> mix, fused message + scatter-add (no mix materialization)
  {
    v16h a0 = frag_a(h_a, 64, row0, 0);
    v16h a1 = frag_a(h_a, 64, row0, 32);
    const int dims[4] = {1, 3, 5, 7};
    const int offs[4] = {0, 64, 256, 576};
    const int yof[4]  = {0, 0, 3, 8};
    #pragma unroll
    for (int t = 0; t < 16; ++t) {
      v8f acc = {};
      acc = wmma_f16(a0, frag_b(w4t, 64, t * 16, 0),  acc);
      acc = wmma_f16(a1, frag_b(w4t, 64, t * 16, 32), acc);
      int li  = t >> 2;
      int ch  = (t & 3) * 16 + n;         // channel 0..63
      int dim = dims[li], off = offs[li], yo = yof[li];
      for (int r = 0; r < 8; ++r) {
        int le = row0 + rb + r;           // local edge index 0..127
        float val = acc[r] * sf_lds[le * 64 + ch];
        float* drow = accum + (size_t)rcv[le] * 1024 + off;
        if (li == 0) {
          unsafeAtomicAdd(drow + ch, val);            // hw global_atomic_add_f32
        } else {
          const float* yl = &y_lds[le * 16 + yo];
          for (int m = 0; m < dim; ++m) unsafeAtomicAdd(drow + ch * dim + m, val * yl[m]);
        }
      }
    }
  }
}

// out[:, off + d*dim + m] = (accum[:, off + c*dim + m] * 0.25) @ W_down[l]; 16 (l,m) GEMMs.
__global__ __launch_bounds__(256) void k_linear_down(
    const float* __restrict__ accum, const _Float16* __restrict__ wdnt, float* __restrict__ out) {
  __shared__ _Float16 a_lds[32 * 64];
  const int dims[4] = {1, 3, 5, 7};
  const int offs[4] = {0, 64, 256, 576};
  int lm = blockIdx.y;
  int l, m;
  if      (lm < 1) { l = 0; m = 0; }
  else if (lm < 4) { l = 1; m = lm - 1; }
  else if (lm < 9) { l = 2; m = lm - 4; }
  else             { l = 3; m = lm - 9; }
  int dim = dims[l], off = offs[l];
  int tid = threadIdx.x;
  int row0g = blockIdx.x * 32;
  for (int i = tid; i < 2048; i += 256) {
    int rr = i >> 6, cc = i & 63;
    int nr = row0g + rr;
    float v = (nr < NN) ? accum[(size_t)nr * 1024 + off + cc * dim + m] * 0.25f : 0.0f;  // * 1/sqrt(16)
    a_lds[i] = (_Float16)v;
  }
  __syncthreads();
  int wid = tid >> 5, rt = wid >> 2, ct = wid & 3;
  const _Float16* bt = wdnt + l * 4096;
  v8f acc = {};
  acc = wmma_f16(frag_a(a_lds, 64, rt * 16, 0),  frag_b(bt, 64, ct * 16, 0),  acc);
  acc = wmma_f16(frag_a(a_lds, 64, rt * 16, 32), frag_b(bt, 64, ct * 16, 32), acc);
  int lane = tid & 31, nc = lane & 15, rb = (lane >> 4) << 3;
  for (int r = 0; r < 8; ++r) {
    int nr = row0g + rt * 16 + rb + r;
    if (nr < NN) out[(size_t)nr * 1024 + off + (size_t)(ct * 16 + nc) * dim + m] = acc[r];
  }
}

extern "C" void kernel_launch(void* const* d_in, const int* in_sizes, int n_in,
                              void* d_out, int out_size, void* d_ws, size_t ws_size,
                              hipStream_t stream) {
  (void)in_sizes; (void)n_in; (void)out_size; (void)ws_size;
  const float* vectors    = (const float*)d_in[0];
  const float* node_feats = (const float*)d_in[1];
  const float* radial     = (const float*)d_in[2];
  const float* W_up       = (const float*)d_in[3];
  const float* w1         = (const float*)d_in[4];
  const float* w2         = (const float*)d_in[5];
  const float* w3         = (const float*)d_in[6];
  const float* w4         = (const float*)d_in[7];
  const float* W_dn       = (const float*)d_in[8];
  const int*   senders    = (const int*)d_in[9];
  const int*   receivers  = (const int*)d_in[10];
  float* out = (float*)d_out;

  // workspace carve-up (f32 then f16 regions; all 16B-aligned)
  float* xup   = (float*)d_ws;                       // [N,64]
  float* accum = xup + (size_t)NN * 64;              // [N,1024]
  _Float16* wupt = (_Float16*)(accum + (size_t)NN * 1024);
  _Float16* w1t  = wupt + 4096;                      // [64][32]
  _Float16* w2t  = w1t + 2048;                       // [64][64]
  _Float16* w3t  = w2t + 4096;                       // [64][64]
  _Float16* w4t  = w3t + 4096;                       // [256][64]
  _Float16* wdnt = w4t + 16384;                      // 4 x [64][64]

  k_zero<<<2048, 256, 0, stream>>>(accum, (long long)NN * 1024);
  k_prep<<<64, 256, 0, stream>>>(W_up, w1, w2, w3, w4, W_dn, wupt, w1t, w2t, w3t, w4t, wdnt);
  k_linear_up<<<(NN + 31) / 32, 256, 0, stream>>>(node_feats, wupt, xup);
  k_edges<<<EE / 128, 256, 0, stream>>>(vectors, radial, senders, receivers, xup,
                                        w1t, w2t, w3t, w4t, accum);
  k_linear_down<<<dim3((NN + 31) / 32, 16), 256, 0, stream>>>(accum, wdnt, out);
}
